// GDAClassifier_10007273800231
// MI455X (gfx1250) — matrix operand order repaired and checked by reading, not compiled, not run on hardware
//
#include <hip/hip_runtime.h>
#include <hip/hip_bf16.h>

// ---------------------------------------------------------------------------
// Problem constants (from the reference)
// ---------------------------------------------------------------------------
#define B_   128
#define T_   256
#define H_   200
#define M_   (B_*T_)      // 32768 rows for the batched input-projection GEMM
#define NP_  832          // 4 gates * 208 (gate dim 200 padded to 208 = 13 tiles)
#define JP_  208
#define K0_  360          // layer-0 input dim (300+30+30)
#define K0P_ 384          // padded to multiple of 32
#define K1_  400          // layer-1 input dim (2H)
#define K1P_ 448
#define KH_  200          // recurrent K
#define KHP_ 224
#define NT_  4            // N-tiles per wave in the batched GEMM (divides 52)

typedef __bf16 bf16_t;
typedef __attribute__((ext_vector_type(16))) __bf16 v16bf;
typedef __attribute__((ext_vector_type(8)))  float  v8f;

// Plain 16-byte chunk (HIP's uint4 has non-trivial ctors -> unusable in union)
struct alignas(16) U4 { unsigned x, y, z, w; };
union Frag { U4 q[2]; v16bf v; };

__device__ __forceinline__ float sigmf(float x) { return 1.0f / (1.0f + __expf(-x)); }
__device__ __forceinline__ bf16_t f2bf(float f) { return (bf16_t)f; }

// ---------------------------------------------------------------------------
// Embedding gather: X0[b*T+t][0..383] (bf16, zero-padded K) + keep mask
// ---------------------------------------------------------------------------
__global__ void embed_gather(const int* __restrict__ words, const int* __restrict__ masks,
                             const int* __restrict__ pos,   const int* __restrict__ ner,
                             const float* __restrict__ emb_w, const float* __restrict__ pos_w,
                             const float* __restrict__ ner_w,
                             bf16_t* __restrict__ X0, float* __restrict__ keep) {
  long i = (long)blockIdx.x * blockDim.x + threadIdx.x;
  long total = (long)M_ * K0P_;
  if (i >= total) return;
  long row = i / K0P_; int k = (int)(i % K0P_);
  float v = 0.0f;
  if (k < 300)      v = emb_w[(long)words[row] * 300 + k];
  else if (k < 330) v = pos_w[pos[row] * 30 + (k - 300)];
  else if (k < 360) v = ner_w[ner[row] * 30 + (k - 330)];
  X0[i] = f2bf(v);
  if (k == 0) keep[row] = (masks[row] == 0) ? 1.0f : 0.0f;
}

// ---------------------------------------------------------------------------
// Weight packing: W[800][K] fp32 -> Wp[832][Kp] bf16, gate-padded + K-padded
// ---------------------------------------------------------------------------
__global__ void pack_w(const float* __restrict__ W, bf16_t* __restrict__ out, int K, int Kp) {
  long i = (long)blockIdx.x * blockDim.x + threadIdx.x;
  long total = (long)NP_ * Kp;
  if (i >= total) return;
  int n = (int)(i / Kp), k = (int)(i % Kp);
  int g = n / JP_, j = n % JP_;
  float v = (j < H_ && k < K) ? W[(long)(g * H_ + j) * K + k] : 0.0f;
  out[i] = f2bf(v);
}

__global__ void pack_bias(const float* __restrict__ b, float* __restrict__ out) {
  int i = blockIdx.x * blockDim.x + threadIdx.x;
  if (i >= NP_) return;
  int g = i / JP_, j = i % JP_;
  out[i] = (j < H_) ? b[g * H_ + j] : 0.0f;
}

// ---------------------------------------------------------------------------
// Batched input projection GEMM (WMMA bf16):
//   Gates[T][B][832](bf16) = A[M][Kp](bf16) * Wp[832][Kp]^T + bias[832]
// Input row m = b*T + t; output row is TIME-MAJOR (t*B + b) so the serial
// recurrence reads one contiguous 212 KB slab per timestep.
// Each wave computes 1 M-tile x NT_ N-tiles: one A-fragment load per k-step
// feeds NT_ WMMAs (4x A reuse). 8 waves / block.
// ---------------------------------------------------------------------------
__global__ __launch_bounds__(256)
void wmma_gemm_bf16(const bf16_t* __restrict__ A, const bf16_t* __restrict__ W,
                    const float* __restrict__ bias, bf16_t* __restrict__ C,
                    int M, int Kp) {
  int wave = threadIdx.x >> 5;
  int lane = threadIdx.x & 31;
  const int groupsN = (NP_ >> 4) / NT_;          // 13
  long grp = (long)blockIdx.x * 8 + wave;
  long nGrps = (long)(M >> 4) * groupsN;
  if (grp >= nGrps) return;                       // wave-uniform guard
  int gn = (int)(grp % groupsN);
  int tm = (int)(grp / groupsN);
  int l15 = lane & 15, lhi = lane >> 4;

  v8f acc[NT_] = {};
  const bf16_t* arow  = A + (long)(tm * 16 + l15) * Kp;
  const bf16_t* wbase = W + (long)(gn * NT_ * 16 + l15) * Kp;
  for (int kb = 0; kb < Kp; kb += 32) {
    Frag af;
    // A fragment: lanes 0-15 hold K [kb..kb+7],[kb+16..kb+23]; lanes 16-31 shifted by 8
    af.q[0] = *(const U4*)(arow + kb + lhi * 8);
    af.q[1] = *(const U4*)(arow + kb + 16 + lhi * 8);
    __builtin_prefetch(arow + kb + 64, 0, 1);
#pragma unroll
    for (int u = 0; u < NT_; ++u) {
      // B fragment: lane l15 = column n; K [kb..kb+7],[kb+16..kb+23], lanes 16-31 shifted by 8
      const bf16_t* wp = wbase + (long)(u * 16) * Kp + kb + lhi * 8;
      Frag bf;
      bf.q[0] = *(const U4*)(wp);
      bf.q[1] = *(const U4*)(wp + 16);
      acc[u] = __builtin_amdgcn_wmma_f32_16x16x32_bf16(false, af.v, false, bf.v,
                                                       (short)0, acc[u], false, false);
    }
  }
#pragma unroll
  for (int u = 0; u < NT_; ++u) {
    int n = (gn * NT_ + u) * 16 + l15;
    float bval = bias[n];
#pragma unroll
    for (int r = 0; r < 8; ++r) {
      int m = tm * 16 + r + 8 * lhi;          // m = b*T + t
      int b = m >> 8, t = m & (T_ - 1);       // T_ == 256
      C[((long)t * B_ + b) * NP_ + n] = f2bf(acc[u][r] + bval);
    }
  }
}

// ---------------------------------------------------------------------------
// LSTM recurrence, BOTH directions of one layer concurrently:
// grid.x == 2, blockIdx.x = direction (0=fwd, 1=bwd), one WGP each.
// State h,c (fp32) + double-buffered bf16 h live in dynamic LDS (312 KB).
// Per step: 104 (m,j) patches x 4 gates, each a K=224 WMMA chain.
//   Gx: precomputed x@W_ih^T + b   [T][B][832] bf16 (time-major)
//   Whh: packed bf16 [832][224]
// ---------------------------------------------------------------------------
#define RECUR_LDS ((B_*H_*4)*2 + 2*(B_*KHP_*2))   // 319488 bytes

__global__ __launch_bounds__(1024)
void lstm_recur(const bf16_t* __restrict__ Gx_f, const bf16_t* __restrict__ Gx_b,
                const bf16_t* __restrict__ Whh_f, const bf16_t* __restrict__ Whh_b,
                const float* __restrict__ keep,
                float* __restrict__ O_f, float* __restrict__ O_b,
                float* __restrict__ hT_f, float* __restrict__ hT_b) {
  const int dir = blockIdx.x;                  // block-uniform
  const bf16_t* __restrict__ Gx  = dir ? Gx_b  : Gx_f;
  const bf16_t* __restrict__ Whh = dir ? Whh_b : Whh_f;
  float* __restrict__ O  = dir ? O_b  : O_f;
  float* __restrict__ hT = dir ? hT_b : hT_f;

  extern __shared__ char smem[];
  float*  hF  = (float*)smem;                 // [128][200]
  float*  cF  = hF + B_ * H_;                 // [128][200]
  bf16_t* hbf = (bf16_t*)(cF + B_ * H_);      // 2 x [128][224]
  const int HB = B_ * KHP_;

  int tid = threadIdx.x;
  for (int i = tid; i < B_ * H_; i += 1024) { hF[i] = 0.0f; cF[i] = 0.0f; }
  for (int i = tid; i < 2 * HB; i += 1024) hbf[i] = f2bf(0.0f);
  __syncthreads();

  int wave = tid >> 5, lane = tid & 31;
  int l15 = lane & 15, lhi = lane >> 4;

  for (int s = 0; s < T_; ++s) {
    int t = dir ? (T_ - 1 - s) : s;
    const bf16_t* __restrict__ Gt = Gx + (long)t * B_ * NP_;  // one 212KB slab
    bf16_t* hcur = hbf + (s & 1) * HB;
    bf16_t* hnx  = hbf + ((s + 1) & 1) * HB;

    for (int p = wave; p < 104; p += 32) {      // 8 m-tiles x 13 j-tiles
      int tm = p & 7, tj = p >> 3;
      v8f acc[4];
#pragma unroll
      for (int g = 0; g < 4; ++g) {
        int col = g * JP_ + tj * 16 + l15;
#pragma unroll
        for (int r = 0; r < 8; ++r) {
          int m = tm * 16 + r + 8 * lhi;
          acc[g][r] = (float)Gt[(long)m * NP_ + col];
        }
      }
      const bf16_t* hrow = hcur + (tm * 16 + l15) * KHP_;
      for (int kb = 0; kb < KHP_; kb += 32) {
        Frag af;
        af.q[0] = *(const U4*)(hrow + kb + lhi * 8);
        af.q[1] = *(const U4*)(hrow + kb + 16 + lhi * 8);
#pragma unroll
        for (int g = 0; g < 4; ++g) {
          const bf16_t* wp = Whh + (long)(g * JP_ + tj * 16 + l15) * KHP_ + kb + lhi * 8;
          Frag bf;
          bf.q[0] = *(const U4*)(wp);
          bf.q[1] = *(const U4*)(wp + 16);
          acc[g] = __builtin_amdgcn_wmma_f32_16x16x32_bf16(false, af.v, false, bf.v,
                                                           (short)0, acc[g], false, false);
        }
      }
      // elementwise cell update (torch gate order i,f,g,o)
      int j = tj * 16 + l15;
#pragma unroll
      for (int r = 0; r < 8; ++r) {
        int m = tm * 16 + r + 8 * lhi;
        float km = keep[(long)m * T_ + t];
        float iv = sigmf(acc[0][r]);
        float fv = sigmf(acc[1][r]);
        float gv = tanhf(acc[2][r]);
        float ov = sigmf(acc[3][r]);
        float cp = 0.0f, hp = 0.0f;
        if (j < H_) { cp = cF[m * H_ + j]; hp = hF[m * H_ + j]; }
        float cn  = fv * cp + iv * gv;
        float hn  = ov * tanhf(cn);
        float hne = km * hn + (1.0f - km) * hp;
        float cne = km * cn + (1.0f - km) * cp;
        if (j < H_) {
          cF[m * H_ + j] = cne;
          hF[m * H_ + j] = hne;
          hnx[m * KHP_ + j] = f2bf(hne);       // pad cols [200..223] stay zero
          O[((long)m * T_ + t) * H_ + j] = hne * km;
        }
      }
    }
    __syncthreads();
  }
  for (int i = tid; i < B_ * H_; i += 1024) hT[i] = hF[i];
}

// ---------------------------------------------------------------------------
// Layer-1 input: X1 = concat(of0, ob0) -> bf16 [M][448], zero-padded
// ---------------------------------------------------------------------------
__global__ void build_x1(const float* __restrict__ O0f, const float* __restrict__ O0b,
                         bf16_t* __restrict__ X1) {
  long i = (long)blockIdx.x * blockDim.x + threadIdx.x;
  long total = (long)M_ * K1P_;
  if (i >= total) return;
  long row = i / K1P_; int k = (int)(i % K1P_);
  float v = 0.0f;
  if (k < H_)          v = O0f[row * H_ + k];
  else if (k < 2 * H_) v = O0b[row * H_ + (k - H_)];
  X1[i] = f2bf(v);
}

// ---------------------------------------------------------------------------
// Span pooling + global-state query assembly: Q[0]=obj_h Q[1]=subj_h Q[2]=glob_h
// ---------------------------------------------------------------------------
__global__ void span_pool(const int* __restrict__ masks, const int* __restrict__ subj_pos,
                          const int* __restrict__ obj_pos,
                          const float* __restrict__ O1f, const float* __restrict__ O1b,
                          const float* __restrict__ hf1, const float* __restrict__ hb1,
                          float* __restrict__ Q) {
  int b = blockIdx.x;
  for (int d = threadIdx.x; d < 2 * H_; d += blockDim.x) {
    float so = 0.0f, ss = 0.0f;
    for (int t = 0; t < T_; ++t) {
      float v = (d < H_) ? O1f[((long)b * T_ + t) * H_ + d]
                         : O1b[((long)b * T_ + t) * H_ + (d - H_)];
      int mk = masks[b * T_ + t];
      if (subj_pos[b * T_ + t] + mk == 0) ss += v;
      if (obj_pos[b * T_ + t] + mk == 0)  so += v;
    }
    Q[(0L * B_ + b) * 2 * H_ + d] = so;
    Q[(1L * B_ + b) * 2 * H_ + d] = ss;
    Q[(2L * B_ + b) * 2 * H_ + d] = (d < H_) ? hb1[b * H_ + d] : hf1[b * H_ + (d - H_)];
  }
}

// ---------------------------------------------------------------------------
// Single-query attention pooling over rnn_out, blockIdx.y selects query
// ---------------------------------------------------------------------------
__global__ __launch_bounds__(256)
void attn(const float* __restrict__ Q, const float* __restrict__ keep,
          const float* __restrict__ O1f, const float* __restrict__ O1b,
          float* __restrict__ Aout) {
  int b = blockIdx.x, qi = blockIdx.y, tid = threadIdx.x;
  __shared__ float qs[2 * H_];
  __shared__ float sc[T_];
  __shared__ float tmp[T_];
  for (int d = tid; d < 2 * H_; d += 256) qs[d] = Q[((long)qi * B_ + b) * 2 * H_ + d];
  __syncthreads();

  // score for timestep t == tid (T_ == 256 == blockDim.x)
  const float* pf = O1f + ((long)b * T_ + tid) * H_;
  const float* pb = O1b + ((long)b * T_ + tid) * H_;
  float dot = 0.0f;
  for (int d = 0; d < H_; ++d) dot += qs[d] * pf[d];
  for (int d = 0; d < H_; ++d) dot += qs[H_ + d] * pb[d];
  float score = (keep[b * T_ + tid] == 0.0f) ? -1e9f : dot * 0.05f;  // 1/sqrt(400)
  tmp[tid] = score;
  __syncthreads();
  for (int s2 = 128; s2 > 0; s2 >>= 1) {
    if (tid < s2) tmp[tid] = fmaxf(tmp[tid], tmp[tid + s2]);
    __syncthreads();
  }
  float mx = tmp[0];
  __syncthreads();
  float e = __expf(score - mx);
  sc[tid] = e; tmp[tid] = e;
  __syncthreads();
  for (int s2 = 128; s2 > 0; s2 >>= 1) {
    if (tid < s2) tmp[tid] += tmp[tid + s2];
    __syncthreads();
  }
  float inv = 1.0f / tmp[0];

  for (int d = tid; d < 2 * H_; d += 256) {
    float a = 0.0f;
    for (int t2 = 0; t2 < T_; ++t2) {
      float v = (d < H_) ? O1f[((long)b * T_ + t2) * H_ + d]
                         : O1b[((long)b * T_ + t2) * H_ + (d - H_)];
      a += sc[t2] * v;
    }
    Aout[((long)qi * B_ + b) * 2 * H_ + d] = a * inv;
  }
}

// ---------------------------------------------------------------------------
// Final head: h = relu(obj_a@Wo^T+bo + subj_a@Ws^T+bs + glob_a@Wg^T+bg); logits
// ---------------------------------------------------------------------------
__global__ __launch_bounds__(256)
void head(const float* __restrict__ Apool,
          const float* __restrict__ wo_w, const float* __restrict__ wo_b,
          const float* __restrict__ ws_w, const float* __restrict__ ws_b,
          const float* __restrict__ wg_w, const float* __restrict__ wg_b,
          const float* __restrict__ cls_w, const float* __restrict__ cls_b,
          float* __restrict__ out) {
  int b = blockIdx.x, tid = threadIdx.x;
  __shared__ float hs[H_];
  const float* a0 = Apool + (0L * B_ + b) * 2 * H_;  // obj_a
  const float* a1 = Apool + (1L * B_ + b) * 2 * H_;  // subj_a
  const float* a2 = Apool + (2L * B_ + b) * 2 * H_;  // glob_a
  if (tid < H_) {
    float acc = wo_b[tid] + ws_b[tid] + wg_b[tid];
    for (int k = 0; k < 2 * H_; ++k)
      acc += a0[k] * wo_w[tid * 2 * H_ + k]
           + a1[k] * ws_w[tid * 2 * H_ + k]
           + a2[k] * wg_w[tid * 2 * H_ + k];
    hs[tid] = fmaxf(acc, 0.0f);
  }
  __syncthreads();
  if (tid < 2) {
    float acc = cls_b[tid];
    for (int j = 0; j < H_; ++j) acc += hs[j] * cls_w[tid * H_ + j];
    out[b * 2 + tid] = acc;
  }
}

// ---------------------------------------------------------------------------
// Host orchestration
// ---------------------------------------------------------------------------
extern "C" void kernel_launch(void* const* d_in, const int* in_sizes, int n_in,
                              void* d_out, int out_size, void* d_ws, size_t ws_size,
                              hipStream_t stream) {
  (void)in_sizes; (void)n_in; (void)out_size; (void)ws_size;
  // --- inputs (setup_inputs dict order) ---
  const int*   words    = (const int*)d_in[0];
  const int*   masks    = (const int*)d_in[1];
  const int*   pos      = (const int*)d_in[2];
  const int*   ner      = (const int*)d_in[3];
  const int*   subj_pos = (const int*)d_in[4];
  const int*   obj_pos  = (const int*)d_in[5];
  const float* emb_w    = (const float*)d_in[6];
  const float* pos_w    = (const float*)d_in[7];
  const float* ner_w    = (const float*)d_in[8];
  const float* w_ih[2][2] = {{(const float*)d_in[9],  (const float*)d_in[12]},
                             {(const float*)d_in[15], (const float*)d_in[18]}};
  const float* w_hh[2][2] = {{(const float*)d_in[10], (const float*)d_in[13]},
                             {(const float*)d_in[16], (const float*)d_in[19]}};
  const float* b_l[2][2]  = {{(const float*)d_in[11], (const float*)d_in[14]},
                             {(const float*)d_in[17], (const float*)d_in[20]}};
  const float* wo_w = (const float*)d_in[21]; const float* wo_b = (const float*)d_in[22];
  const float* ws_w = (const float*)d_in[23]; const float* ws_b = (const float*)d_in[24];
  const float* wg_w = (const float*)d_in[25]; const float* wg_b = (const float*)d_in[26];
  const float* cls_w = (const float*)d_in[27]; const float* cls_b = (const float*)d_in[28];
  float* out = (float*)d_out;

  // --- workspace layout (all offsets 256B aligned) ---
  char* ws = (char*)d_ws;
  size_t off = 0;
  bf16_t* X0 = (bf16_t*)(ws + off);       off += (size_t)M_ * K0P_ * 2;   // 25.2 MB
  bf16_t* X1 = (bf16_t*)(ws + off);       off += (size_t)M_ * K1P_ * 2;   // 29.4 MB
  bf16_t* Wih[2][2];
  Wih[0][0] = (bf16_t*)(ws + off); off += (size_t)NP_ * K0P_ * 2;
  Wih[0][1] = (bf16_t*)(ws + off); off += (size_t)NP_ * K0P_ * 2;
  Wih[1][0] = (bf16_t*)(ws + off); off += (size_t)NP_ * K1P_ * 2;
  Wih[1][1] = (bf16_t*)(ws + off); off += (size_t)NP_ * K1P_ * 2;
  bf16_t* Whh[2][2];
  for (int l = 0; l < 2; ++l)
    for (int d = 0; d < 2; ++d) { Whh[l][d] = (bf16_t*)(ws + off); off += (size_t)NP_ * KHP_ * 2; }
  float* bias[2][2];
  for (int l = 0; l < 2; ++l)
    for (int d = 0; d < 2; ++d) { bias[l][d] = (float*)(ws + off); off += (size_t)NP_ * 4; }
  float* keep = (float*)(ws + off);       off += (size_t)M_ * 4;
  bf16_t* GxF = (bf16_t*)(ws + off);      off += (size_t)M_ * NP_ * 2;    // 54.5 MB (time-major)
  bf16_t* GxB = (bf16_t*)(ws + off);      off += (size_t)M_ * NP_ * 2;    // 54.5 MB (time-major)
  float* O0f  = (float*)(ws + off);       off += (size_t)M_ * H_ * 4;     // 26.2 MB
  float* O0b  = (float*)(ws + off);       off += (size_t)M_ * H_ * 4;
  float* O1f  = O0f;                      // reuse: X1 already holds layer-0 outputs
  float* O1b  = O0b;
  float* hf1  = (float*)(ws + off);       off += (size_t)B_ * H_ * 4;
  float* hb1  = (float*)(ws + off);       off += (size_t)B_ * H_ * 4;
  float* Q    = (float*)(ws + off);       off += (size_t)3 * B_ * 2 * H_ * 4;
  float* Ap   = (float*)(ws + off);       off += (size_t)3 * B_ * 2 * H_ * 4;

  // allow 312 KB dynamic LDS for the recurrence kernel
  (void)hipFuncSetAttribute((const void*)lstm_recur,
                            hipFuncAttributeMaxDynamicSharedMemorySize, RECUR_LDS);

  // 1) embedding gather + mask
  {
    long total = (long)M_ * K0P_;
    embed_gather<<<(unsigned)((total + 255) / 256), 256, 0, stream>>>(
        words, masks, pos, ner, emb_w, pos_w, ner_w, X0, keep);
  }
  // 2) pack weights/biases to padded bf16 layouts
  const int Kin[2]  = {K0_, K1_};
  const int KinP[2] = {K0P_, K1P_};
  for (int l = 0; l < 2; ++l)
    for (int d = 0; d < 2; ++d) {
      long ti = (long)NP_ * KinP[l];
      pack_w<<<(unsigned)((ti + 255) / 256), 256, 0, stream>>>(w_ih[l][d], Wih[l][d], Kin[l], KinP[l]);
      long th = (long)NP_ * KHP_;
      pack_w<<<(unsigned)((th + 255) / 256), 256, 0, stream>>>(w_hh[l][d], Whh[l][d], KH_, KHP_);
      pack_bias<<<(NP_ + 255) / 256, 256, 0, stream>>>(b_l[l][d], bias[l][d]);
    }

  const int gemmBlocks = ((M_ / 16) * ((NP_ / 16) / NT_) + 7) / 8;   // 3328

  // 3) layer 0: both direction GEMMs, then both recurrences concurrently
  wmma_gemm_bf16<<<gemmBlocks, 256, 0, stream>>>(X0, Wih[0][0], bias[0][0], GxF, M_, K0P_);
  wmma_gemm_bf16<<<gemmBlocks, 256, 0, stream>>>(X0, Wih[0][1], bias[0][1], GxB, M_, K0P_);
  lstm_recur<<<2, 1024, RECUR_LDS, stream>>>(GxF, GxB, Whh[0][0], Whh[0][1], keep,
                                             O0f, O0b, hf1, hb1);

  // 4) layer-1 input
  {
    long total = (long)M_ * K1P_;
    build_x1<<<(unsigned)((total + 255) / 256), 256, 0, stream>>>(O0f, O0b, X1);
  }

  // 5) layer 1 (outputs alias layer-0 output buffers; X1 holds the data now)
  wmma_gemm_bf16<<<gemmBlocks, 256, 0, stream>>>(X1, Wih[1][0], bias[1][0], GxF, M_, K1P_);
  wmma_gemm_bf16<<<gemmBlocks, 256, 0, stream>>>(X1, Wih[1][1], bias[1][1], GxB, M_, K1P_);
  lstm_recur<<<2, 1024, RECUR_LDS, stream>>>(GxF, GxB, Whh[1][0], Whh[1][1], keep,
                                             O1f, O1b, hf1, hb1);

  // 6) pooling heads
  span_pool<<<B_, 256, 0, stream>>>(masks, subj_pos, obj_pos, O1f, O1b, hf1, hb1, Q);
  attn<<<dim3(B_, 3), 256, 0, stream>>>(Q, keep, O1f, O1b, Ap);
  head<<<B_, 256, 0, stream>>>(Ap, wo_w, wo_b, ws_w, ws_b, wg_w, wg_b, cls_w, cls_b, out);
}
